// ImageCaptionModel_30485677867917
// MI455X (gfx1250) — matrix operand order repaired and checked by reading, not compiled, not run on hardware
//
#include <hip/hip_runtime.h>

// ---------------------------------------------------------------------------
// CDNA5 (gfx1250) ImageCaption GRU decoder.
//   - bf16 WMMA (v_wmma_f32_16x16x32_bf16) everywhere, fp32 accumulate
//   - persistent grid-synced kernel for the sequential 40-step recurrence
//     (waves of a block share the weight stream -> WGP$ broadcast reuse)
//   - big [5120,512]x[512,10000] projection with LDS-staged B operands via
//     CDNA5 async global->LDS loads (double buffered), 8x B reuse per block
// ---------------------------------------------------------------------------

typedef __bf16 v16bf __attribute__((ext_vector_type(16)));
typedef __bf16 v8bf  __attribute__((ext_vector_type(8)));
typedef float  v8f   __attribute__((ext_vector_type(8)));
typedef int    vi4   __attribute__((vector_size(16)));   // builtin's b128 unit

union BFrag { v16bf v; v8bf h[2]; };

#define BB   128
#define TT   40
#define HID  512
#define IN0  1024
#define VOC  10000

// ---------------- CDNA5 async global->LDS staging (guarded) ----------------
#if __has_builtin(__builtin_amdgcn_global_load_async_to_lds_b128) && \
    __has_builtin(__builtin_amdgcn_s_wait_asynccnt)
#define USE_ASYNC_LDS 1
#endif

__device__ __forceinline__ void stage16(const __bf16* g, __bf16* l) {
#ifdef USE_ASYNC_LDS
  __builtin_amdgcn_global_load_async_to_lds_b128(
      (__attribute__((address_space(1))) vi4*)g,
      (__attribute__((address_space(3))) vi4*)l, 0, 0);
#else
  *(v8bf*)l = *(const v8bf*)g;   // sync fallback: global load + ds_store
#endif
}
__device__ __forceinline__ void stage_wait() {
#ifdef USE_ASYNC_LDS
  __builtin_amdgcn_s_wait_asynccnt(0);
#endif
}

// ---------------------------------------------------------------------------
// Grid-wide barrier: per-use counter slot, release add + acquire spin.
// Only valid when all blocks are co-resident (we launch 32 blocks).
// ---------------------------------------------------------------------------
__device__ __forceinline__ void gbar(unsigned* bar, int idx, unsigned nblk) {
  __syncthreads();
  if (threadIdx.x == 0) {
    __hip_atomic_fetch_add(&bar[idx], 1u, __ATOMIC_RELEASE, __HIP_MEMORY_SCOPE_AGENT);
    while (__hip_atomic_load(&bar[idx], __ATOMIC_ACQUIRE, __HIP_MEMORY_SCOPE_AGENT) < nblk) {
      __builtin_amdgcn_s_sleep(1);
    }
  }
  __syncthreads();
  __threadfence();   // acquire: invalidate stale near-cache lines
}

// ---------------------------------------------------------------------------
// 16x16 fp32 tile accumulation over nk K-steps of 32 (bf16 WMMA).
// A: row-major bf16 activations (16-bit A layout, ISA 7.12.2):
//    lane m=lane&15, K=(e>>3)*16+(lane>>4)*8+(e&7) -> two 16B loads/lane.
// B: packed tiles Wp[(kt*ntiles+nt)*32+lane][e], N=lane&15, K=(e>>3)*16+(lane>>4)*8+(e&7)
//    -> one contiguous 32B per lane.
// NOTE: weight pointers intentionally NOT __restrict__ and the inner loop
// uses a bounded unroll: the weight stream is invariant across the 40-step
// outer loop, and full hoisting previously spilled hundreds of fragments to
// scratch. Bounded unroll keeps live B fragments in-register.
// ---------------------------------------------------------------------------
__device__ __forceinline__ void mma_seg(v8f& acc,
    const __bf16* __restrict__ Abase, int lda, int mrow, int k0, int nk,
    const __bf16* Wp, int ntiles, int nt, int ktbase, int lane)
{
  const int half = lane >> 4;
  const int ml   = lane & 15;
  const __bf16* ap = Abase + (size_t)(mrow + ml) * lda + k0 + half * 8;
  const __bf16* bp = Wp + (((size_t)ktbase * ntiles + nt) * 32 + lane) * 16;
  const size_t bstride = (size_t)ntiles * 512;
#pragma unroll 4
  for (int ks = 0; ks < nk; ++ks) {
    BFrag a, b;
    a.h[0] = *(const v8bf*)(ap);
    a.h[1] = *(const v8bf*)(ap + 16);
    b.h[0] = *(const v8bf*)(bp);
    b.h[1] = *(const v8bf*)(bp + 8);
    acc = __builtin_amdgcn_wmma_f32_16x16x32_bf16(false, a.v, false, b.v,
                                                  (short)0, acc, false, false);
    ap += 32;
    bp += bstride;
  }
}

// Dual-gate variant: one A fragment feeds two weight streams (z and r gates).
__device__ __forceinline__ void mma_dual(v8f& acc0, v8f& acc1,
    const __bf16* __restrict__ Abase, int lda, int mrow, int k0, int nk,
    const __bf16* W0, const __bf16* W1,
    int ntiles, int nt, int ktbase, int lane)
{
  const int half = lane >> 4;
  const int ml   = lane & 15;
  const __bf16* ap = Abase + (size_t)(mrow + ml) * lda + k0 + half * 8;
  const size_t boff = (((size_t)ktbase * ntiles + nt) * 32 + lane) * 16;
  const __bf16* bp0 = W0 + boff;
  const __bf16* bp1 = W1 + boff;
  const size_t bstride = (size_t)ntiles * 512;
#pragma unroll 4
  for (int ks = 0; ks < nk; ++ks) {
    BFrag a, b0, b1;
    a.h[0]  = *(const v8bf*)(ap);
    a.h[1]  = *(const v8bf*)(ap + 16);
    b0.h[0] = *(const v8bf*)(bp0);
    b0.h[1] = *(const v8bf*)(bp0 + 8);
    b1.h[0] = *(const v8bf*)(bp1);
    b1.h[1] = *(const v8bf*)(bp1 + 8);
    acc0 = __builtin_amdgcn_wmma_f32_16x16x32_bf16(false, a.v, false, b0.v,
                                                   (short)0, acc0, false, false);
    acc1 = __builtin_amdgcn_wmma_f32_16x16x32_bf16(false, a.v, false, b1.v,
                                                   (short)0, acc1, false, false);
    ap += 32;
    bp0 += bstride;
    bp1 += bstride;
  }
}

// ---------------------------------------------------------------------------
__global__ __launch_bounds__(256) void zero_k(unsigned* bar, float* h0, float* h1,
                                              __bf16* h0b, __bf16* h1b) {
  int i = blockIdx.x * 256 + threadIdx.x;
  if (i < 256) bar[i] = 0u;
  if (i < BB * HID) {
    h0[i] = 0.0f; h1[i] = 0.0f;
    h0b[i] = (__bf16)0.0f; h1b[i] = (__bf16)0.0f;
  }
}

// Embedding gather + CNN concat -> bf16 step inputs xb[t][b][0..1023]
__global__ __launch_bounds__(256) void embed_k(const int* __restrict__ tokens,
                                               const float* __restrict__ cnn,
                                               const float* __restrict__ emb,
                                               __bf16* __restrict__ xb) {
  int t = blockIdx.x / BB, b = blockIdx.x % BB;
  int tok = tokens[b * TT + t];
  __bf16* dst = xb + ((size_t)t * BB + b) * IN0;
  const float* e = emb + (size_t)tok * 512;
  const float* c = cnn + (size_t)b * 512;
  for (int i = threadIdx.x; i < 512; i += 256) {
    dst[i]       = (__bf16)e[i];
    dst[512 + i] = (__bf16)c[i];
  }
}

// Pack fp32 weight [K,N] row-major -> bf16 WMMA B-operand tiles.
__global__ __launch_bounds__(256) void pack_w(const float* __restrict__ W,
                                              __bf16* __restrict__ Wp,
                                              int K, int N) {
  int idx = blockIdx.x * 256 + threadIdx.x;
  if (idx >= K * N) return;
  int e    = idx & 15;
  int lane = (idx >> 4) & 31;
  int tile = idx >> 9;
  int ntiles = N >> 4;
  int nt = tile % ntiles, kt = tile / ntiles;
  int k = kt * 32 + (e >> 3) * 16 + (lane >> 4) * 8 + (e & 7);
  int n = nt * 16 + (lane & 15);
  Wp[idx] = (__bf16)W[(size_t)k * N + n];
}

// ---------------------------------------------------------------------------
// Persistent sequential GRU. 32 blocks x 8 waves; wave w owns M-tile w
// (8 tiles = all 128 rows), block owns N-tile nt = blockIdx.x, so all 8
// waves share one weight stream (WGP$ broadcast reuse).
// ---------------------------------------------------------------------------
__global__ __launch_bounds__(256) void gru_seq(
    const __bf16* __restrict__ xb,
    const __bf16* wu0p, const __bf16* wr0p, const __bf16* wc0p,
    const __bf16* wu1p, const __bf16* wr1p, const __bf16* wc1p,
    const float* __restrict__ bu0, const float* __restrict__ br0,
    const float* __restrict__ bc0, const float* __restrict__ bu1,
    const float* __restrict__ br1, const float* __restrict__ bc1,
    float* h0, float* h1, __bf16* h0b, __bf16* h1b,
    float* zws, __bf16* rhb, __bf16* h1ball,
    float* hidden_out, unsigned* bar)
{
  const int lane = threadIdx.x & 31;
  const int widx = threadIdx.x >> 5;           // wave index = M-tile
  const unsigned nblk = gridDim.x;             // 32
  const int nt   = blockIdx.x;                 // N-tile (0..31)
  const int half = lane >> 4, nl = lane & 15;
  const int mrow = widx * 16;
  int barIdx = 0;

#pragma unroll 1
  for (int t = 0; t < TT; ++t) {
    // Opaque barrier: forbid hoisting the (t-invariant) weight-fragment
    // loads out of the timestep loop -- that previously spilled to scratch.
    asm volatile("" ::: "memory");

    const __bf16* xb_t = xb + (size_t)t * BB * IN0;

    // ---- Layer 0, phase A: z and r gates (fused, shared A fragments) ----
    {
      v8f aU = {}, aR = {};
      mma_dual(aU, aR, h0b, 512, mrow, 0, 16, wu0p, wr0p, 32, nt, 0, lane);
      mma_dual(aU, aR, xb_t, IN0, mrow, 0, 32, wu0p, wr0p, 32, nt, 16, lane);
      for (int r = 0; r < 8; ++r) {
        int m = mrow + r + 8 * half;
        int n = nt * 16 + nl;
        int idx = m * 512 + n;
        float z = 1.0f / (1.0f + __expf(-(aU[r] + bu0[n])));
        float g = 1.0f / (1.0f + __expf(-(aR[r] + br0[n])));
        zws[idx] = z;
        rhb[idx] = (__bf16)(g * h0[idx]);
      }
    }
    gbar(bar, barIdx++, nblk);

    // ---- Layer 0, phase B: candidate + combine --------------------------
    {
      v8f acc = {};
      mma_seg(acc, rhb, 512, mrow, 0, 16, wc0p, 32, nt, 0, lane);
      mma_seg(acc, xb_t, IN0, mrow, 0, 32, wc0p, 32, nt, 16, lane);
      for (int r = 0; r < 8; ++r) {
        int m = mrow + r + 8 * half;
        int n = nt * 16 + nl;
        int idx = m * 512 + n;
        float hh = tanhf(acc[r] + bc0[n]);
        float z  = zws[idx];
        float hn = z * h0[idx] + (1.0f - z) * hh;
        h0[idx]  = hn;
        h0b[idx] = (__bf16)hn;
      }
    }
    gbar(bar, barIdx++, nblk);

    // ---- Layer 1, phase A ----------------------------------------------
    {
      v8f aU = {}, aR = {};
      mma_dual(aU, aR, h1b, 512, mrow, 0, 16, wu1p, wr1p, 32, nt, 0, lane);
      mma_dual(aU, aR, h0b, 512, mrow, 0, 16, wu1p, wr1p, 32, nt, 16, lane);
      for (int r = 0; r < 8; ++r) {
        int m = mrow + r + 8 * half;
        int n = nt * 16 + nl;
        int idx = m * 512 + n;
        float z = 1.0f / (1.0f + __expf(-(aU[r] + bu1[n])));
        float g = 1.0f / (1.0f + __expf(-(aR[r] + br1[n])));
        zws[idx] = z;
        rhb[idx] = (__bf16)(g * h1[idx]);
      }
    }
    gbar(bar, barIdx++, nblk);

    // ---- Layer 1, phase B ----------------------------------------------
    {
      v8f acc = {};
      mma_seg(acc, rhb, 512, mrow, 0, 16, wc1p, 32, nt, 0, lane);
      mma_seg(acc, h0b, 512, mrow, 0, 16, wc1p, 32, nt, 16, lane);
      for (int r = 0; r < 8; ++r) {
        int m = mrow + r + 8 * half;
        int n = nt * 16 + nl;
        int idx = m * 512 + n;
        float hh = tanhf(acc[r] + bc1[n]);
        float z  = zws[idx];
        float hn = z * h1[idx] + (1.0f - z) * hh;
        h1[idx]  = hn;
        h1b[idx] = (__bf16)hn;
        h1ball[(size_t)t * (BB * HID) + idx] = (__bf16)hn;  // row t*128+b
      }
    }
    gbar(bar, barIdx++, nblk);
  }

  // final hidden state: stack([h0, h1]) -> tail of d_out
  int tid = blockIdx.x * blockDim.x + threadIdx.x;
  int tot = gridDim.x * blockDim.x;
  for (int i = tid; i < BB * HID; i += tot) {
    hidden_out[i]            = h0[i];
    hidden_out[BB * HID + i] = h1[i];
  }
}

// ---------------------------------------------------------------------------
// Output projection: [5120,512](bf16) x [512,10000](bf16) -> fp32 logits.
// Block = 8 M-tiles (one per wave) x 4 N-tiles. B tiles staged in LDS with
// double-buffered async global->LDS loads; all 8 waves share them.
// ---------------------------------------------------------------------------
#define NGRP 157
__global__ __launch_bounds__(256) void proj_k(const __bf16* __restrict__ h1ball,
                                              const __bf16* __restrict__ owp,
                                              const float* __restrict__ outb,
                                              float* __restrict__ out) {
  __shared__ __align__(16) __bf16 bs[2][4][512];
  const int tid  = threadIdx.x;
  const int lane = tid & 31;
  const int w    = tid >> 5;
  const int mg   = blockIdx.x / NGRP;        // 0..39
  const int ng   = blockIdx.x % NGRP;
  const int mt   = mg * 8 + w;               // 0..319
  const int nt0  = ng * 4;
  const int nv   = (625 - nt0 < 4) ? (625 - nt0) : 4;
  const int half = lane >> 4, ml = lane & 15;

  // staging role: thread -> (tile jst, 16B chunk c); clamp dead tiles
  const int jst = tid >> 6;
  const int c   = tid & 63;
  const int jcl = (jst < nv) ? jst : (nv - 1);

  stage16(owp + ((size_t)(nt0 + jcl) * 512) + c * 8, &bs[0][jst][c * 8]);
  stage_wait();
  __syncthreads();

  v8f acc[4] = {};
  const __bf16* ap = h1ball + (size_t)(mt * 16 + ml) * 512 + half * 8;
  for (int ks = 0; ks < 16; ++ks) {
    if (ks + 1 < 16) {
      stage16(owp + ((size_t)((ks + 1) * 625 + nt0 + jcl) * 512) + c * 8,
              &bs[(ks + 1) & 1][jst][c * 8]);
    }
    BFrag a;
    a.h[0] = *(const v8bf*)(ap);
    a.h[1] = *(const v8bf*)(ap + 16);
    for (int j = 0; j < nv; ++j) {
      BFrag b;
      const __bf16* lp = &bs[ks & 1][j][lane * 16];
      b.h[0] = *(const v8bf*)(lp);
      b.h[1] = *(const v8bf*)(lp + 8);
      acc[j] = __builtin_amdgcn_wmma_f32_16x16x32_bf16(false, a.v, false, b.v,
                                                       (short)0, acc[j], false, false);
    }
    ap += 32;
    stage_wait();
    __syncthreads();
  }

  for (int j = 0; j < nv; ++j) {
    int n = (nt0 + j) * 16 + ml;
    float bias = outb[n];
    for (int r = 0; r < 8; ++r) {
      int m  = mt * 16 + r + 8 * half;     // row = t*128 + b
      int tt = m >> 7, bb = m & 127;
      out[((size_t)bb * TT + tt) * VOC + n] = acc[j][r] + bias;
    }
  }
}

// ---------------------------------------------------------------------------
extern "C" void kernel_launch(void* const* d_in, const int* in_sizes, int n_in,
                              void* d_out, int out_size, void* d_ws, size_t ws_size,
                              hipStream_t stream) {
  const int*   tokens = (const int*)  d_in[0];
  const float* cnn    = (const float*)d_in[1];
  const float* emb    = (const float*)d_in[2];
  const float* wu0 = (const float*)d_in[3];  const float* bu0 = (const float*)d_in[4];
  const float* wr0 = (const float*)d_in[5];  const float* br0 = (const float*)d_in[6];
  const float* wc0 = (const float*)d_in[7];  const float* bc0 = (const float*)d_in[8];
  const float* wu1 = (const float*)d_in[9];  const float* bu1 = (const float*)d_in[10];
  const float* wr1 = (const float*)d_in[11]; const float* br1 = (const float*)d_in[12];
  const float* wc1 = (const float*)d_in[13]; const float* bc1 = (const float*)d_in[14];
  const float* ow  = (const float*)d_in[15]; const float* ob  = (const float*)d_in[16];
  float* out = (float*)d_out;

  char* ws = (char*)d_ws;
  size_t off = 0;
  auto alloc = [&](size_t bytes) -> void* {
    off = (off + 255) & ~(size_t)255;
    void* p = ws + off;
    off += bytes;
    return p;
  };

  unsigned* bar   = (unsigned*)alloc(256 * sizeof(unsigned));
  float*    zws   = (float*)   alloc((size_t)BB * HID * 4);
  float*    h0    = (float*)   alloc((size_t)BB * HID * 4);
  float*    h1    = (float*)   alloc((size_t)BB * HID * 4);
  __bf16*   h0b   = (__bf16*)  alloc((size_t)BB * HID * 2);
  __bf16*   h1b   = (__bf16*)  alloc((size_t)BB * HID * 2);
  __bf16*   rhb   = (__bf16*)  alloc((size_t)BB * HID * 2);
  __bf16*   xb    = (__bf16*)  alloc((size_t)TT * BB * IN0 * 2);
  __bf16*   h1all = (__bf16*)  alloc((size_t)TT * BB * HID * 2);
  __bf16*   wu0p  = (__bf16*)  alloc((size_t)1536 * 512 * 2);
  __bf16*   wr0p  = (__bf16*)  alloc((size_t)1536 * 512 * 2);
  __bf16*   wc0p  = (__bf16*)  alloc((size_t)1536 * 512 * 2);
  __bf16*   wu1p  = (__bf16*)  alloc((size_t)1024 * 512 * 2);
  __bf16*   wr1p  = (__bf16*)  alloc((size_t)1024 * 512 * 2);
  __bf16*   wc1p  = (__bf16*)  alloc((size_t)1024 * 512 * 2);
  __bf16*   owp   = (__bf16*)  alloc((size_t)512 * VOC * 2);

  zero_k<<<256, 256, 0, stream>>>(bar, h0, h1, h0b, h1b);
  embed_k<<<TT * BB, 256, 0, stream>>>(tokens, cnn, emb, xb);

  pack_w<<<(1536 * 512 + 255) / 256, 256, 0, stream>>>(wu0, wu0p, 1536, 512);
  pack_w<<<(1536 * 512 + 255) / 256, 256, 0, stream>>>(wr0, wr0p, 1536, 512);
  pack_w<<<(1536 * 512 + 255) / 256, 256, 0, stream>>>(wc0, wc0p, 1536, 512);
  pack_w<<<(1024 * 512 + 255) / 256, 256, 0, stream>>>(wu1, wu1p, 1024, 512);
  pack_w<<<(1024 * 512 + 255) / 256, 256, 0, stream>>>(wr1, wr1p, 1024, 512);
  pack_w<<<(1024 * 512 + 255) / 256, 256, 0, stream>>>(wc1, wc1p, 1024, 512);
  pack_w<<<(512 * VOC + 255) / 256, 256, 0, stream>>>(ow, owp, 512, VOC);

  float* hidden_out = out + (size_t)BB * TT * VOC;
  gru_seq<<<32, 256, 0, stream>>>(xb, wu0p, wr0p, wc0p, wu1p, wr1p, wc1p,
                                  bu0, br0, bc0, bu1, br1, bc1,
                                  h0, h1, h0b, h1b, zws, rhb, h1all,
                                  hidden_out, bar);

  proj_k<<<40 * NGRP, 256, 0, stream>>>(h1all, owp, ob, out);
}
